// AttentionPairBias_11708080849025
// MI455X (gfx1250) — matrix operand order repaired and checked by reading, not compiled, not run on hardware
//
#include <hip/hip_runtime.h>
#include <hip/hip_bf16.h>

// ---------------------------------------------------------------------------
// AttentionPairBias for MI455X (gfx1250): bf16 WMMA GEMMs + VALU LN kernels.
// Sizes: B=1, N=768, A=768, S=384, H=16, D=48, P=128.
// Dual-tile waves (1 A-frag -> 2 WMMAs) + capped unroll to avoid spills.
// ---------------------------------------------------------------------------

typedef __attribute__((ext_vector_type(16))) __bf16 v16bf;
typedef __attribute__((ext_vector_type(8)))  float  v8f;

#define NTOK 768
#define ADIM 768
#define SDIM 384
#define PDIM 128
#define HN   16
#define DHD  48
#define DPAD 64

__device__ __forceinline__ float sigmoidf(float x) { return 1.0f / (1.0f + __expf(-x)); }

// --- WMMA fragment loaders (wave32, 16x16x32 bf16). ------------------------
// A (16x32, MxK): lanes 0-15 row M=lane, K=kb..kb+7 and kb+16..kb+23;
//                 lanes 16-31 row M=lane-16, K=kb+8..kb+15 and kb+24..kb+31.
__device__ __forceinline__ v16bf load_a_frag(const __bf16* A, int lda, int mbase,
                                             int kbase, int lane) {
  const __bf16* p = A + (size_t)(mbase + (lane & 15)) * lda + kbase + ((lane >> 4) << 3);
  union { uint4 u[2]; v16bf v; } r;
  r.u[0] = *reinterpret_cast<const uint4*>(p);       // K: koff .. koff+7
  r.u[1] = *reinterpret_cast<const uint4*>(p + 16);  // K: koff+16 .. koff+23
  return r.v;
}

// B (32x16, KxN) supplied as Bt[N][K] row-major: lane n = lane&15,
// K = kb + 8*(lane>>4) + 0..7 and kb + 16 + 8*(lane>>4) + 0..7 (matches A).
__device__ __forceinline__ v16bf load_b_frag(const __bf16* Bt, int ldb, int nbase,
                                             int kbase, int lane) {
  const __bf16* p = Bt + (size_t)(nbase + (lane & 15)) * ldb + kbase + ((lane >> 4) << 3);
  union { uint4 u[2]; v16bf v; } r;
  r.u[0] = *reinterpret_cast<const uint4*>(p);
  r.u[1] = *reinterpret_cast<const uint4*>(p + 16);
  return r.v;
}

__device__ __forceinline__ v8f wmma_bf16(v16bf a, v16bf b, v8f c) {
  return __builtin_amdgcn_wmma_f32_16x16x32_bf16(false, a, false, b, (short)0, c,
                                                 false, false);
}

// Single-tile K loop.
__device__ __forceinline__ v8f wmma_loop(const __bf16* A, int lda, const __bf16* Bt,
                                         int ldb, int mbase, int nbase, int K,
                                         int lane, v8f acc) {
#pragma unroll 2
  for (int kb = 0; kb < K; kb += 32) {
    if (kb + 32 < K) {
      __builtin_prefetch(A + (size_t)(mbase + (lane & 15)) * lda + kb + 32, 0, 1);
      __builtin_prefetch(Bt + (size_t)(nbase + (lane & 15)) * ldb + kb + 32, 0, 1);
    }
    v16bf af = load_a_frag(A, lda, mbase, kb, lane);
    v16bf bfr = load_b_frag(Bt, ldb, nbase, kb, lane);
    acc = wmma_bf16(af, bfr, acc);
  }
  return acc;
}

// Dual-tile K loop: one shared A fragment feeds two B streams / two accs.
__device__ __forceinline__ void wmma_loop2(const __bf16* A, int lda,
                                           const __bf16* B0, int ldb0, int n0,
                                           const __bf16* B1, int ldb1, int n1,
                                           int mbase, int K, int lane,
                                           v8f& acc0, v8f& acc1) {
#pragma unroll 2
  for (int kb = 0; kb < K; kb += 32) {
    if (kb + 32 < K) {
      __builtin_prefetch(A + (size_t)(mbase + (lane & 15)) * lda + kb + 32, 0, 1);
      __builtin_prefetch(B0 + (size_t)(n0 + (lane & 15)) * ldb0 + kb + 32, 0, 1);
      __builtin_prefetch(B1 + (size_t)(n1 + (lane & 15)) * ldb1 + kb + 32, 0, 1);
    }
    v16bf af = load_a_frag(A, lda, mbase, kb, lane);
    v16bf b0 = load_b_frag(B0, ldb0, n0, kb, lane);
    v16bf b1 = load_b_frag(B1, ldb1, n1, kb, lane);
    acc0 = wmma_bf16(af, b0, acc0);
    acc1 = wmma_bf16(af, b1, acc1);
  }
}

// C/D 16x16 f32: VGPR r -> row mbase + r + 8*(lane>>4), col nbase + (lane&15).

// --- small utility kernels --------------------------------------------------
__global__ void k_transpose_bf16(const float* __restrict__ src, __bf16* __restrict__ dst,
                                 int R, int C) {
  int idx = blockIdx.x * 256 + threadIdx.x;
  if (idx < R * C) {
    int r = idx / C, c = idx % C;
    dst[(size_t)c * R + r] = (__bf16)src[idx];
  }
}

__global__ void k_zero_bf16(__bf16* __restrict__ p, int n) {
  int idx = blockIdx.x * 256 + threadIdx.x;
  if (idx < n) p[idx] = (__bf16)0.0f;
}

// --- LayerNorm kernels ------------------------------------------------------
__global__ void k_ln_a(const float* __restrict__ a, float* __restrict__ a_n) {
  __shared__ float buf[256];
  int row = blockIdx.x, tid = threadIdx.x;
  const float* x = a + (size_t)row * ADIM;
  float v0 = x[tid], v1 = x[tid + 256], v2 = x[tid + 512];
  buf[tid] = v0 + v1 + v2; __syncthreads();
  for (int s = 128; s > 0; s >>= 1) { if (tid < s) buf[tid] += buf[tid + s]; __syncthreads(); }
  float mean = buf[0] * (1.0f / 768.0f); __syncthreads();
  float d0 = v0 - mean, d1 = v1 - mean, d2 = v2 - mean;
  buf[tid] = d0 * d0 + d1 * d1 + d2 * d2; __syncthreads();
  for (int s = 128; s > 0; s >>= 1) { if (tid < s) buf[tid] += buf[tid + s]; __syncthreads(); }
  float rstd = rsqrtf(buf[0] * (1.0f / 768.0f) + 1e-5f);
  float* o = a_n + (size_t)row * ADIM;
  o[tid] = d0 * rstd; o[tid + 256] = d1 * rstd; o[tid + 512] = d2 * rstd;
}

__global__ void k_ln_s(const float* __restrict__ s, const float* __restrict__ ln_s_w,
                       __bf16* __restrict__ s_ln, __bf16* __restrict__ s_raw) {
  __shared__ float buf[256];
  int row = blockIdx.x, tid = threadIdx.x;
  const float* x = s + (size_t)row * SDIM;
  float v0 = x[tid];
  float v1 = (tid < 128) ? x[tid + 256] : 0.0f;
  buf[tid] = v0 + v1; __syncthreads();
  for (int st = 128; st > 0; st >>= 1) { if (tid < st) buf[tid] += buf[tid + st]; __syncthreads(); }
  float mean = buf[0] * (1.0f / 384.0f); __syncthreads();
  float d0 = v0 - mean, d1 = v1 - mean;
  buf[tid] = d0 * d0 + ((tid < 128) ? d1 * d1 : 0.0f); __syncthreads();
  for (int st = 128; st > 0; st >>= 1) { if (tid < st) buf[tid] += buf[tid + st]; __syncthreads(); }
  float rstd = rsqrtf(buf[0] * (1.0f / 384.0f) + 1e-5f);
  s_ln[(size_t)row * SDIM + tid] = (__bf16)(d0 * rstd * ln_s_w[tid]);
  s_raw[(size_t)row * SDIM + tid] = (__bf16)v0;
  if (tid < 128) {
    s_ln[(size_t)row * SDIM + 256 + tid] = (__bf16)(d1 * rstd * ln_s_w[256 + tid]);
    s_raw[(size_t)row * SDIM + 256 + tid] = (__bf16)v1;
  }
}

// --- AdaLN: a2 = sigmoid(s_n@gamma + gb) * a_n + s_n@shift ------------------
// Dual-B: one A fragment feeds both gamma and shift GEMMs.
__global__ void k_gemm_a2(const __bf16* __restrict__ s_ln, const __bf16* __restrict__ gammaT,
                          const __bf16* __restrict__ shiftT, const float* __restrict__ gamma_b,
                          const float* __restrict__ a_n, __bf16* __restrict__ a2) {
  int wave = blockIdx.x * 8 + (threadIdx.x >> 5);
  int lane = threadIdx.x & 31;
  int mt = wave / (ADIM / 16), nt = wave % (ADIM / 16);
  v8f accg = {}; v8f accs = {};
  wmma_loop2(s_ln, SDIM, gammaT, SDIM, nt * 16, shiftT, SDIM, nt * 16,
             mt * 16, SDIM, lane, accg, accs);
  for (int r = 0; r < 8; ++r) {
    int row = mt * 16 + r + ((lane >> 4) << 3);
    int col = nt * 16 + (lane & 15);
    float sg = sigmoidf(accg[r] + gamma_b[col]);
    float val = sg * a_n[(size_t)row * ADIM + col] + accs[r];
    a2[(size_t)row * ADIM + col] = (__bf16)val;
  }
}

// --- QKV + gate: 4 GEMMs a2[768,768] @ W^T (dual-N per wave) ----------------
__global__ void k_gemm_qkvg(const __bf16* __restrict__ a2, const __bf16* __restrict__ WqT,
                            const __bf16* __restrict__ WkT, const __bf16* __restrict__ WvT,
                            const __bf16* __restrict__ WgT, const float* __restrict__ bq,
                            __bf16* __restrict__ qp, __bf16* __restrict__ kp,
                            __bf16* __restrict__ vT, float* __restrict__ g) {
  int which = blockIdx.y;
  int wave = blockIdx.x * 8 + (threadIdx.x >> 5);
  int lane = threadIdx.x & 31;
  int mt = wave / 24, ntp = wave % 24;          // 48 m-tiles x 24 n-tile-pairs
  int n0 = ntp * 32, n1 = ntp * 32 + 16;
  const __bf16* Wt = (which == 0) ? WqT : (which == 1) ? WkT : (which == 2) ? WvT : WgT;
  v8f acc0 = {}; v8f acc1 = {};
  wmma_loop2(a2, ADIM, Wt, ADIM, n0, Wt, ADIM, n1, mt * 16, ADIM, lane, acc0, acc1);
  const float rs = 0.14433756729740643f;  // 1/sqrt(48), folded into q
  for (int t = 0; t < 2; ++t) {
    v8f acc = t ? acc1 : acc0;
    int nb = t ? n1 : n0;
    for (int r = 0; r < 8; ++r) {
      int row = mt * 16 + r + ((lane >> 4) << 3);
      int col = nb + (lane & 15);
      int h = col / DHD, d = col % DHD;
      float v = acc[r];
      if (which == 0)      qp[((size_t)h * NTOK + row) * DPAD + d] = (__bf16)((v + bq[col]) * rs);
      else if (which == 1) kp[((size_t)h * NTOK + row) * DPAD + d] = (__bf16)v;
      else if (which == 2) vT[((size_t)h * DHD + d) * NTOK + row] = (__bf16)v;
      else                 g[((size_t)h * NTOK + row) * DHD + d] = sigmoidf(v);
    }
  }
}

// --- pair bias: scores[h][i][j] = LN(pair[i,j,:])@Wb[:,h] + beta[i][j] ------
__global__ void k_bias(const float* __restrict__ pair, const float* __restrict__ beta,
                       const float* __restrict__ lnpw, const float* __restrict__ lnpb,
                       const float* __restrict__ Wb, float* __restrict__ scores) {
  __shared__ float wbs[PDIM * HN];
  __shared__ float yb[8][PDIM];
  int tid = threadIdx.x;
  for (int i = tid; i < PDIM * HN; i += 256) wbs[i] = Wb[i];
  __syncthreads();
  int wave = tid >> 5, lane = tid & 31;
  int pid = blockIdx.x * 8 + wave;
  int i = pid / NTOK, j = pid % NTOK;
  const float* x = pair + ((size_t)i * NTOK + j) * PDIM;
  float x0 = x[lane], x1 = x[lane + 32], x2 = x[lane + 64], x3 = x[lane + 96];
  float s1 = x0 + x1 + x2 + x3;
  float s2 = x0 * x0 + x1 * x1 + x2 * x2 + x3 * x3;
  for (int m = 16; m > 0; m >>= 1) { s1 += __shfl_xor(s1, m); s2 += __shfl_xor(s2, m); }
  float mean = s1 * (1.0f / 128.0f);
  float var = s2 * (1.0f / 128.0f) - mean * mean;
  float rstd = rsqrtf(var + 1e-5f);
  yb[wave][lane]      = (x0 - mean) * rstd * lnpw[lane]      + lnpb[lane];
  yb[wave][lane + 32] = (x1 - mean) * rstd * lnpw[lane + 32] + lnpb[lane + 32];
  yb[wave][lane + 64] = (x2 - mean) * rstd * lnpw[lane + 64] + lnpb[lane + 64];
  yb[wave][lane + 96] = (x3 - mean) * rstd * lnpw[lane + 96] + lnpb[lane + 96];
  __syncthreads();
  int h = lane & 15, half = lane >> 4;
  float dot = 0.0f;
  for (int c = 0; c < 64; ++c) dot += yb[wave][half * 64 + c] * wbs[(half * 64 + c) * HN + h];
  dot += __shfl_xor(dot, 16);
  if (lane < 16)
    scores[((size_t)h * NTOK + i) * NTOK + j] = dot + beta[(size_t)i * NTOK + j];
}

// --- scores += q @ k^T (bias already resident; dual key-tiles per wave) -----
__global__ void k_scores(const __bf16* __restrict__ qp, const __bf16* __restrict__ kp,
                         float* __restrict__ scores) {
  int id = blockIdx.x * 8 + (threadIdx.x >> 5);
  int lane = threadIdx.x & 31;
  int h = id / (48 * 24);
  int rem = id % (48 * 24);
  int mt = rem / 24, ntp = rem % 24;
  int n0 = ntp * 32, n1 = ntp * 32 + 16;
  float* sc = scores + (size_t)h * NTOK * NTOK;
  v8f acc0, acc1;
  for (int r = 0; r < 8; ++r) {
    int row = mt * 16 + r + ((lane >> 4) << 3);
    int col = lane & 15;
    acc0[r] = sc[(size_t)row * NTOK + n0 + col];
    acc1[r] = sc[(size_t)row * NTOK + n1 + col];
  }
  const __bf16* q = qp + (size_t)h * NTOK * DPAD;
  const __bf16* k = kp + (size_t)h * NTOK * DPAD;
  wmma_loop2(q, DPAD, k, DPAD, n0, k, DPAD, n1, mt * 16, DPAD, lane, acc0, acc1);
  for (int r = 0; r < 8; ++r) {
    int row = mt * 16 + r + ((lane >> 4) << 3);
    int col = lane & 15;
    sc[(size_t)row * NTOK + n0 + col] = acc0[r];
    sc[(size_t)row * NTOK + n1 + col] = acc1[r];
  }
}

// --- softmax over keys, emit bf16 attn --------------------------------------
__global__ void k_softmax(const float* __restrict__ scores, __bf16* __restrict__ attn) {
  __shared__ float buf[256];
  int row = blockIdx.x;  // h*768 + q
  int tid = threadIdx.x;
  const float* x = scores + (size_t)row * NTOK;
  float v0 = x[tid], v1 = x[tid + 256], v2 = x[tid + 512];
  buf[tid] = fmaxf(v0, fmaxf(v1, v2)); __syncthreads();
  for (int s = 128; s > 0; s >>= 1) { if (tid < s) buf[tid] = fmaxf(buf[tid], buf[tid + s]); __syncthreads(); }
  float mx = buf[0]; __syncthreads();
  float e0 = __expf(v0 - mx), e1 = __expf(v1 - mx), e2 = __expf(v2 - mx);
  buf[tid] = e0 + e1 + e2; __syncthreads();
  for (int s = 128; s > 0; s >>= 1) { if (tid < s) buf[tid] += buf[tid + s]; __syncthreads(); }
  float inv = 1.0f / buf[0];
  __bf16* o = attn + (size_t)row * NTOK;
  o[tid] = (__bf16)(e0 * inv); o[tid + 256] = (__bf16)(e1 * inv); o[tid + 512] = (__bf16)(e2 * inv);
}

// --- o[q, h*48+d] = (attn[h] @ v[h]) * g ------------------------------------
__global__ void k_av(const __bf16* __restrict__ attn, const __bf16* __restrict__ vT,
                     const float* __restrict__ g, __bf16* __restrict__ o) {
  int id = blockIdx.x * 8 + (threadIdx.x >> 5);
  int lane = threadIdx.x & 31;
  int h = id / (48 * 3);
  int rem = id % (48 * 3);
  int mt = rem / 3, nt = rem % 3;
  v8f acc = {};
  acc = wmma_loop(attn + (size_t)h * NTOK * NTOK, NTOK,
                  vT + (size_t)h * DHD * NTOK, NTOK,
                  mt * 16, nt * 16, NTOK, lane, acc);
  for (int r = 0; r < 8; ++r) {
    int row = mt * 16 + r + ((lane >> 4) << 3);
    int d = nt * 16 + (lane & 15);
    float val = acc[r] * g[((size_t)h * NTOK + row) * DHD + d];
    o[(size_t)row * ADIM + h * DHD + d] = (__bf16)val;
  }
}

// --- gate_out = sigmoid(s @ Wout + bout), dual-N ----------------------------
__global__ void k_gate(const __bf16* __restrict__ s_raw, const __bf16* __restrict__ WoutT,
                       const float* __restrict__ bout, float* __restrict__ gate_out) {
  int wave = blockIdx.x * 8 + (threadIdx.x >> 5);
  int lane = threadIdx.x & 31;
  int mt = wave / 24, ntp = wave % 24;
  int n0 = ntp * 32, n1 = ntp * 32 + 16;
  v8f acc0 = {}; v8f acc1 = {};
  wmma_loop2(s_raw, SDIM, WoutT, SDIM, n0, WoutT, SDIM, n1, mt * 16, SDIM, lane,
             acc0, acc1);
  for (int r = 0; r < 8; ++r) {
    int row = mt * 16 + r + ((lane >> 4) << 3);
    int col0 = n0 + (lane & 15), col1 = n1 + (lane & 15);
    gate_out[(size_t)row * ADIM + col0] = sigmoidf(acc0[r] + bout[col0]);
    gate_out[(size_t)row * ADIM + col1] = sigmoidf(acc1[r] + bout[col1]);
  }
}

// --- out = gate_out * (o @ Wp), dual-N --------------------------------------
__global__ void k_out(const __bf16* __restrict__ o, const __bf16* __restrict__ WpT,
                      const float* __restrict__ gate_out, float* __restrict__ out) {
  int wave = blockIdx.x * 8 + (threadIdx.x >> 5);
  int lane = threadIdx.x & 31;
  int mt = wave / 24, ntp = wave % 24;
  int n0 = ntp * 32, n1 = ntp * 32 + 16;
  v8f acc0 = {}; v8f acc1 = {};
  wmma_loop2(o, ADIM, WpT, ADIM, n0, WpT, ADIM, n1, mt * 16, ADIM, lane, acc0, acc1);
  for (int r = 0; r < 8; ++r) {
    int row = mt * 16 + r + ((lane >> 4) << 3);
    int col0 = n0 + (lane & 15), col1 = n1 + (lane & 15);
    out[(size_t)row * ADIM + col0] = acc0[r] * gate_out[(size_t)row * ADIM + col0];
    out[(size_t)row * ADIM + col1] = acc1[r] * gate_out[(size_t)row * ADIM + col1];
  }
}

// ---------------------------------------------------------------------------
extern "C" void kernel_launch(void* const* d_in, const int* in_sizes, int n_in,
                              void* d_out, int out_size, void* d_ws, size_t ws_size,
                              hipStream_t stream) {
  (void)in_sizes; (void)n_in; (void)out_size; (void)ws_size;
  const float* a       = (const float*)d_in[0];
  const float* s       = (const float*)d_in[1];
  const float* pair    = (const float*)d_in[2];
  const float* beta    = (const float*)d_in[3];
  const float* ln_s_w  = (const float*)d_in[4];
  const float* gamma_w = (const float*)d_in[5];
  const float* gamma_b = (const float*)d_in[6];
  const float* shift_w = (const float*)d_in[7];
  const float* Wq      = (const float*)d_in[8];
  const float* bq      = (const float*)d_in[9];
  const float* Wk      = (const float*)d_in[10];
  const float* Wv      = (const float*)d_in[11];
  const float* ln_p_w  = (const float*)d_in[12];
  const float* ln_p_b  = (const float*)d_in[13];
  const float* Wb      = (const float*)d_in[14];
  const float* Wg      = (const float*)d_in[15];
  const float* Wp      = (const float*)d_in[16];
  const float* Wout    = (const float*)d_in[17];
  const float* bout    = (const float*)d_in[18];
  float* out = (float*)d_out;

  char* ws = (char*)d_ws;
  size_t off = 0;
  auto take = [&](size_t bytes) -> void* {
    void* p = ws + off;
    off += (bytes + 255) & ~(size_t)255;
    return p;
  };
  float*  a_n    = (float*) take((size_t)NTOK * ADIM * 4);
  __bf16* s_ln   = (__bf16*)take((size_t)NTOK * SDIM * 2);
  __bf16* s_raw  = (__bf16*)take((size_t)NTOK * SDIM * 2);
  __bf16* a2     = (__bf16*)take((size_t)NTOK * ADIM * 2);
  __bf16* gammaT = (__bf16*)take((size_t)ADIM * SDIM * 2);
  __bf16* shiftT = (__bf16*)take((size_t)ADIM * SDIM * 2);
  __bf16* WqT    = (__bf16*)take((size_t)ADIM * ADIM * 2);
  __bf16* WkT    = (__bf16*)take((size_t)ADIM * ADIM * 2);
  __bf16* WvT    = (__bf16*)take((size_t)ADIM * ADIM * 2);
  __bf16* WgT    = (__bf16*)take((size_t)ADIM * ADIM * 2);
  __bf16* WpT    = (__bf16*)take((size_t)ADIM * ADIM * 2);
  __bf16* WoutT  = (__bf16*)take((size_t)ADIM * SDIM * 2);
  __bf16* qp     = (__bf16*)take((size_t)HN * NTOK * DPAD * 2);
  __bf16* kp     = (__bf16*)take((size_t)HN * NTOK * DPAD * 2);
  __bf16* vT     = (__bf16*)take((size_t)HN * DHD * NTOK * 2);
  float*  g      = (float*) take((size_t)HN * NTOK * DHD * 4);
  float*  scores = (float*) take((size_t)HN * NTOK * NTOK * 4);
  __bf16* attn   = (__bf16*)take((size_t)HN * NTOK * NTOK * 2);
  __bf16* obuf   = (__bf16*)take((size_t)NTOK * ADIM * 2);
  float*  gate_o = (float*) take((size_t)NTOK * ADIM * 4);

  // 1. weight transposes -> bf16 (Bt layouts)
  k_transpose_bf16<<<(SDIM * ADIM + 255) / 256, 256, 0, stream>>>(gamma_w, gammaT, SDIM, ADIM);
  k_transpose_bf16<<<(SDIM * ADIM + 255) / 256, 256, 0, stream>>>(shift_w, shiftT, SDIM, ADIM);
  k_transpose_bf16<<<(ADIM * ADIM + 255) / 256, 256, 0, stream>>>(Wq, WqT, ADIM, ADIM);
  k_transpose_bf16<<<(ADIM * ADIM + 255) / 256, 256, 0, stream>>>(Wk, WkT, ADIM, ADIM);
  k_transpose_bf16<<<(ADIM * ADIM + 255) / 256, 256, 0, stream>>>(Wv, WvT, ADIM, ADIM);
  k_transpose_bf16<<<(ADIM * ADIM + 255) / 256, 256, 0, stream>>>(Wg, WgT, ADIM, ADIM);
  k_transpose_bf16<<<(ADIM * ADIM + 255) / 256, 256, 0, stream>>>(Wp, WpT, ADIM, ADIM);
  k_transpose_bf16<<<(SDIM * ADIM + 255) / 256, 256, 0, stream>>>(Wout, WoutT, SDIM, ADIM);

  // 2. zero the D-padding of q/k (cols 48..63 stay zero)
  k_zero_bf16<<<(HN * NTOK * DPAD + 255) / 256, 256, 0, stream>>>(qp, HN * NTOK * DPAD);
  k_zero_bf16<<<(HN * NTOK * DPAD + 255) / 256, 256, 0, stream>>>(kp, HN * NTOK * DPAD);

  // 3. LayerNorms
  k_ln_a<<<NTOK, 256, 0, stream>>>(a, a_n);
  k_ln_s<<<NTOK, 256, 0, stream>>>(s, ln_s_w, s_ln, s_raw);

  // 4. AdaLN -> a2 (bf16)
  k_gemm_a2<<<(48 * 48) / 8, 256, 0, stream>>>(s_ln, gammaT, shiftT, gamma_b, a_n, a2);

  // 5. QKV + gate (dual-N waves: 48 x 24 tile-pairs)
  k_gemm_qkvg<<<dim3((48 * 24) / 8, 4), 256, 0, stream>>>(a2, WqT, WkT, WvT, WgT, bq,
                                                          qp, kp, vT, g);

  // 6. pair bias -> scores (one wave32 per (i,j))
  k_bias<<<(NTOK * NTOK) / 8, 256, 0, stream>>>(pair, beta, ln_p_w, ln_p_b, Wb, scores);

  // 7. scores += q @ k^T (scale folded into q; dual key-tiles)
  k_scores<<<(HN * 48 * 24) / 8, 256, 0, stream>>>(qp, kp, scores);

  // 8. softmax -> attn (bf16)
  k_softmax<<<HN * NTOK, 256, 0, stream>>>(scores, attn);

  // 9. o = (attn @ v) * g
  k_av<<<(HN * 48 * 3) / 8, 256, 0, stream>>>(attn, vT, g, obuf);

  // 10. final gate + output projection (dual-N)
  k_gate<<<(48 * 24) / 8, 256, 0, stream>>>(s_raw, WoutT, bout, gate_o);
  k_out<<<(48 * 24) / 8, 256, 0, stream>>>(obuf, WpT, gate_o, out);
}
